// MultiHeadedAttention_68212670595959
// MI455X (gfx1250) — matrix operand order repaired and checked by reading, not compiled, not run on hardware
//
#include <hip/hip_runtime.h>

#define B_ 2
#define S_ 2048
#define D_ 1024
#define H_ 16
#define DH_ 64
#define MROWS (B_*S_)   // 4096

typedef __attribute__((ext_vector_type(16))) _Float16 v16h;
typedef __attribute__((ext_vector_type(8)))  _Float16 v8h;
typedef __attribute__((ext_vector_type(8)))  float    v8f;
typedef __attribute__((ext_vector_type(4)))  float    f32x4;
typedef __attribute__((ext_vector_type(4)))  int      i32x4;

union V16H { v16h v; v8h h[2]; };
union V8H  { v8h v; _Float16 e[8]; };

#define GLOBAL_AS __attribute__((address_space(1)))
#define LDS_AS    __attribute__((address_space(3)))

#if defined(__gfx1250__) && __has_builtin(__builtin_amdgcn_global_load_async_to_lds_b128)
#define HAVE_ASYNC_LDS 1
#else
#define HAVE_ASYNC_LDS 0
#endif

#if HAVE_ASYNC_LDS
# if __has_builtin(__builtin_amdgcn_s_wait_asynccnt)
#  define WAIT_ASYNC(n) __builtin_amdgcn_s_wait_asynccnt(n)
# else
#  define WAIT_ASYNC(n) asm volatile("s_wait_asynccnt %0" :: "i"(n) : "memory")
# endif
#else
# define WAIT_ASYNC(n) ((void)0)
#endif

// Copy 16 bytes (8 halves) global -> LDS. Async (ASYNCcnt) when available.
__device__ inline void cp16_g2l(const _Float16* g, _Float16* l) {
#if HAVE_ASYNC_LDS
  __builtin_amdgcn_global_load_async_to_lds_b128(
      (GLOBAL_AS i32x4*)(g), (LDS_AS i32x4*)(l), 0, 0);
#else
  *(v8h*)l = *(const v8h*)g;
#endif
}

// Load a 16-half fragment: halves [0..7] at p, halves [8..15] at p+gap.
// gap=16 -> A and B fragments (K 0..7 / 16..23 per lane-half)
__device__ inline v16h frag_ab(const _Float16* p, int gap) {
  V16H u;
  u.h[0] = *(const v8h*)p;
  u.h[1] = *(const v8h*)(p + gap);
  return u.v;
}

// ---------------------------------------------------------------------------
// One-time f32 -> f16 conversion (streaming, non-temporal reads).
// ---------------------------------------------------------------------------
__global__ __launch_bounds__(256) void cvt_f32_f16(
    const float* __restrict__ in, _Float16* __restrict__ out, int n8) {
  const int i = blockIdx.x * 256 + threadIdx.x;
  if (i >= n8) return;
  const f32x4* p = (const f32x4*)in + (size_t)i * 2;
  f32x4 a = __builtin_nontemporal_load(p);
  f32x4 b = __builtin_nontemporal_load(p + 1);
  V8H u;
#pragma unroll
  for (int j = 0; j < 4; j++) { u.e[j] = (_Float16)a[j]; u.e[4+j] = (_Float16)b[j]; }
  *(v8h*)(out + (size_t)i * 8) = u.v;
}

// ---------------------------------------------------------------------------
// Y[M=4096, N=1024] = A[M,K=1024] @ W[N,K]^T + bias[N]   (A, W already f16)
// 128x128 workgroup tile, 8 waves (32x64 each), K-stage 64 (2 WMMA k-steps).
// Double-buffered LDS fed by async global->LDS copies.
// OUTF32: write f32 (final projection) else f16 (feeds attention / next GEMM).
// ---------------------------------------------------------------------------
template <bool OUTF32>
__global__ __launch_bounds__(256) void gemm_f16wmma(
    const _Float16* __restrict__ A, const _Float16* __restrict__ W,
    const float* __restrict__ bias, float* __restrict__ Yf,
    _Float16* __restrict__ Yh) {
  const int N = D_, K = D_;
  __shared__ _Float16 ldsA[2][128 * 72];   // 128 rows x 64 halves (+8 pad = 144B stride)
  __shared__ _Float16 ldsB[2][128 * 72];

  const int tid    = threadIdx.x;
  const int lane   = tid & 31;
  const int lane16 = lane & 15;
  const int lo8    = (lane & 16) ? 8 : 0;
  const int lo16   = (lane & 16) ? 16 : 0;
  const int wave   = tid >> 5;
  const int waveM  = wave & 3;
  const int waveN  = wave >> 2;
  const int m0 = blockIdx.y * 128;
  const int n0 = blockIdx.x * 128;

  // tile copy mapping: 128 rows x 64 halves, 2 threads/row, 4 x 16B each
  const int cpRow = tid >> 1;
  const int cpCol = (tid & 1) * 32;

  const _Float16* gA = A + (size_t)(m0 + cpRow) * K + cpCol;
  const _Float16* gW = W + (size_t)(n0 + cpRow) * K + cpCol;

  auto issue = [&](int kt, int buf) {
    _Float16* lA = &ldsA[buf][cpRow * 72 + cpCol];
    _Float16* lB = &ldsB[buf][cpRow * 72 + cpCol];
#pragma unroll
    for (int j = 0; j < 4; j++) cp16_g2l(gA + kt + j * 8, lA + j * 8);
#pragma unroll
    for (int j = 0; j < 4; j++) cp16_g2l(gW + kt + j * 8, lB + j * 8);
  };

  v8f acc[2][4] = {};

  const int T = K / 64;          // 16 stages
  issue(0, 0);
  for (int t = 0; t < T; ++t) {
    const int cur = t & 1;
    if (t + 1 < T) { issue((t + 1) * 64, cur ^ 1); WAIT_ASYNC(8); }
    else           { WAIT_ASYNC(0); }
    __syncthreads();

#pragma unroll
    for (int kk = 0; kk < 2; kk++) {
      v16h aF[2], bF[4];
#pragma unroll
      for (int mi = 0; mi < 2; mi++)
        aF[mi] = frag_ab(&ldsA[cur][(waveM * 32 + mi * 16 + lane16) * 72 + kk * 32 + lo8], 16);
#pragma unroll
      for (int ni = 0; ni < 4; ni++)
        bF[ni] = frag_ab(&ldsB[cur][(waveN * 64 + ni * 16 + lane16) * 72 + kk * 32 + lo8], 16);
#pragma unroll
      for (int mi = 0; mi < 2; mi++)
#pragma unroll
        for (int ni = 0; ni < 4; ni++)
          acc[mi][ni] = __builtin_amdgcn_wmma_f32_16x16x32_f16(
              false, aF[mi], false, bF[ni], (short)0, acc[mi][ni], false, false);
    }
    __syncthreads();
  }

#pragma unroll
  for (int mi = 0; mi < 2; mi++) {
#pragma unroll
    for (int ni = 0; ni < 4; ni++) {
      const int col = n0 + waveN * 64 + ni * 16 + lane16;
      const float bv = bias[col];
#pragma unroll
      for (int r = 0; r < 8; r++) {
        const int row = m0 + waveM * 32 + mi * 16 + r + lo8;
        const float v = acc[mi][ni][r] + bv;
        if (OUTF32) Yf[(size_t)row * N + col] = v;
        else        Yh[(size_t)row * N + col] = (_Float16)v;
      }
    }
  }
}

// ---------------------------------------------------------------------------
// Flash attention, causal, f16 in / f16 out (fp32 accumulation + softmax).
// One workgroup = (b, h, 128-query tile); 8 waves x 16 query rows.
// K tiles async-copied to double-buffered LDS; V manually transposed (f16).
// ---------------------------------------------------------------------------
__global__ __launch_bounds__(256) void flash_attn_wmma(
    const _Float16* __restrict__ Qp, const _Float16* __restrict__ Kp,
    const _Float16* __restrict__ Vp, _Float16* __restrict__ Out) {
  __shared__ _Float16 ldsQ[128 * 72];        // 128 x 64 (+8 pad)
  __shared__ _Float16 ldsK[2][64 * 72];      // keys x dh, double buffered
  __shared__ _Float16 ldsVt[2][64 * 72];     // dh x keys (transposed), double buffered
  __shared__ _Float16 ldsP[8 * 16 * 64];     // per-wave P scratch

  const int tid    = threadIdx.x;
  const int lane   = tid & 31;
  const int lane16 = lane & 15;
  const int hi8    = (lane & 16) ? 8 : 0;
  const int lo16   = hi8 * 2;
  const int wq     = tid >> 5;
  const int qBase  = blockIdx.x * 128;
  const int h      = blockIdx.y;
  const int b      = blockIdx.z;

  const _Float16* Qh = Qp + (size_t)b * S_ * D_ + h * DH_;
  const _Float16* Kh = Kp + (size_t)b * S_ * D_ + h * DH_;
  const _Float16* Vh = Vp + (size_t)b * S_ * D_ + h * DH_;

  const int kRow = tid >> 2, kCol = (tid & 3) * 16;   // 64x64 tile mapping

  auto issueK = [&](int kb, int buf) {
    const _Float16* g = Kh + (size_t)(kb * 64 + kRow) * D_ + kCol;
    cp16_g2l(g,     &ldsK[buf][kRow * 72 + kCol]);
    cp16_g2l(g + 8, &ldsK[buf][kRow * 72 + kCol + 8]);
  };
  auto issueV = [&](int kb, int buf) {   // transposed scatter (sync)
    const _Float16* g = Vh + (size_t)(kb * 64 + kRow) * D_ + kCol;
    v8h a = *(const v8h*)g, c = *(const v8h*)(g + 8);
#pragma unroll
    for (int j = 0; j < 8; j++) {
      ldsVt[buf][(kCol + j) * 72 + kRow]     = a[j];
      ldsVt[buf][(kCol + 8 + j) * 72 + kRow] = c[j];
    }
  };

  // Q tile 128x64 (async) + first K/V block
  {
    const int row = tid >> 1, colB = (tid & 1) * 32;
    const _Float16* g = Qh + (size_t)(qBase + row) * D_ + colB;
    _Float16* l = &ldsQ[row * 72 + colB];
#pragma unroll
    for (int j = 0; j < 4; j++) cp16_g2l(g + j * 8, l + j * 8);
  }
  issueK(0, 0);
  issueV(0, 0);
  WAIT_ASYNC(0);
  __syncthreads();

  v16h aQ[2];
#pragma unroll
  for (int kk = 0; kk < 2; kk++)
    aQ[kk] = frag_ab(&ldsQ[(wq * 16 + lane16) * 72 + kk * 32 + hi8], 16);

  float rM[8], rL[8];
  v8f oacc[4] = {};
#pragma unroll
  for (int r = 0; r < 8; r++) { rM[r] = -1e30f; rL[r] = 0.f; }

  const int nkb = (qBase + 128) >> 6;   // causal block limit
  for (int kb = 0; kb < nkb; kb++) {
    const int cur = kb & 1;
    if (kb + 1 < nkb) { issueK(kb + 1, cur ^ 1); issueV(kb + 1, cur ^ 1); WAIT_ASYNC(2); }
    else              { WAIT_ASYNC(0); }
    __syncthreads();

    // scores 16x64 per wave
    v8f sc[4] = {};
#pragma unroll
    for (int ni = 0; ni < 4; ni++) {
#pragma unroll
      for (int kk = 0; kk < 2; kk++) {
        v16h bK = frag_ab(&ldsK[cur][(ni * 16 + lane16) * 72 + kk * 32 + hi8], 16);
        sc[ni] = __builtin_amdgcn_wmma_f32_16x16x32_f16(
            false, aQ[kk], false, bK, (short)0, sc[ni], false, false);
      }
    }

    // scale + causal mask + online softmax (row stats lane-uniform per half)
    const float scale = 0.125f;   // 1/sqrt(64)
#pragma unroll
    for (int r = 0; r < 8; r++) {
      const int qrow = qBase + wq * 16 + r + hi8;
      float rowm = -1e30f;
#pragma unroll
      for (int ni = 0; ni < 4; ni++) {
        const int kcol = kb * 64 + ni * 16 + lane16;
        float s = sc[ni][r] * scale;
        if (kcol > qrow) s = -1e30f;
        sc[ni][r] = s;
        rowm = fmaxf(rowm, s);
      }
#pragma unroll
      for (int m = 8; m >= 1; m >>= 1) rowm = fmaxf(rowm, __shfl_xor(rowm, m, 32));
      const float mnew = fmaxf(rM[r], rowm);
      const float corr = __expf(rM[r] - mnew);
      rM[r] = mnew;
      float psum = 0.f;
#pragma unroll
      for (int ni = 0; ni < 4; ni++) {
        const float p = __expf(sc[ni][r] - mnew);
        sc[ni][r] = p;
        psum += p;
      }
#pragma unroll
      for (int m = 8; m >= 1; m >>= 1) psum += __shfl_xor(psum, m, 32);
      rL[r] = rL[r] * corr + psum;
#pragma unroll
      for (int ni = 0; ni < 4; ni++) oacc[ni][r] *= corr;
    }

    // P (C-layout) -> per-wave LDS scratch (row-major 16x64)
#pragma unroll
    for (int ni = 0; ni < 4; ni++)
#pragma unroll
      for (int r = 0; r < 8; r++)
        ldsP[wq * 1024 + (r + hi8) * 64 + ni * 16 + lane16] = (_Float16)sc[ni][r];

    // O += P @ V
#pragma unroll
    for (int kk = 0; kk < 2; kk++) {
      v16h pA = frag_ab(&ldsP[wq * 1024 + lane16 * 64 + kk * 32 + hi8], 16);
#pragma unroll
      for (int ni = 0; ni < 4; ni++) {
        v16h bV = frag_ab(&ldsVt[cur][(ni * 16 + lane16) * 72 + kk * 32 + hi8], 16);
        oacc[ni] = __builtin_amdgcn_wmma_f32_16x16x32_f16(
            false, pA, false, bV, (short)0, oacc[ni], false, false);
      }
    }
    __syncthreads();
  }

  // normalize + write merged[b, q, h*64 + dh] as f16 (feeds O-projection GEMM)
#pragma unroll
  for (int r = 0; r < 8; r++) {
    const int qrow = qBase + wq * 16 + r + hi8;
    const float inv = 1.f / rL[r];
#pragma unroll
    for (int ni = 0; ni < 4; ni++) {
      const int dh = ni * 16 + lane16;
      Out[((size_t)b * S_ + qrow) * D_ + h * DH_ + dh] = (_Float16)(oacc[ni][r] * inv);
    }
  }
}

extern "C" void kernel_launch(void* const* d_in, const int* in_sizes, int n_in,
                              void* d_out, int out_size, void* d_ws, size_t ws_size,
                              hipStream_t stream) {
  (void)in_sizes; (void)n_in; (void)out_size; (void)ws_size;
  const float* queries = (const float*)d_in[0];
  const float* keys    = (const float*)d_in[1];
  const float* values  = (const float*)d_in[2];
  const float* W_Q     = (const float*)d_in[3];
  const float* b_Q     = (const float*)d_in[4];
  const float* W_K     = (const float*)d_in[5];
  const float* b_K     = (const float*)d_in[6];
  const float* W_V     = (const float*)d_in[7];
  const float* b_V     = (const float*)d_in[8];
  const float* W_O     = (const float*)d_in[9];
  const float* b_O     = (const float*)d_in[10];

  const size_t nBig = (size_t)MROWS * D_;   // 4 Mi elements
  const size_t nW   = (size_t)D_ * D_;      // 1 Mi elements

  _Float16* qh  = (_Float16*)d_ws;          // converted activations
  _Float16* kh  = qh + nBig;
  _Float16* vh  = kh + nBig;
  _Float16* Wqh = vh + nBig;                // converted weights
  _Float16* Wkh = Wqh + nW;
  _Float16* Wvh = Wkh + nW;
  _Float16* Woh = Wvh + nW;
  _Float16* Qp  = Woh + nW;                 // f16 projections
  _Float16* Kp  = Qp + nBig;
  _Float16* Vp  = Kp + nBig;
  _Float16* Mg  = Vp + nBig;                // f16 merged attention output

  const int gBig = (int)(nBig / 8 / 256);   // 2048 blocks
  const int gW   = (int)(nW / 8 / 256);     // 512 blocks
  cvt_f32_f16<<<gBig, 256, 0, stream>>>(queries, qh, (int)(nBig / 8));
  cvt_f32_f16<<<gBig, 256, 0, stream>>>(keys,    kh, (int)(nBig / 8));
  cvt_f32_f16<<<gBig, 256, 0, stream>>>(values,  vh, (int)(nBig / 8));
  cvt_f32_f16<<<gW,   256, 0, stream>>>(W_Q, Wqh, (int)(nW / 8));
  cvt_f32_f16<<<gW,   256, 0, stream>>>(W_K, Wkh, (int)(nW / 8));
  cvt_f32_f16<<<gW,   256, 0, stream>>>(W_V, Wvh, (int)(nW / 8));
  cvt_f32_f16<<<gW,   256, 0, stream>>>(W_O, Woh, (int)(nW / 8));

  dim3 gGemm(D_ / 128, MROWS / 128);
  gemm_f16wmma<false><<<gGemm, 256, 0, stream>>>(qh, Wqh, b_Q, nullptr, Qp);
  gemm_f16wmma<false><<<gGemm, 256, 0, stream>>>(kh, Wkh, b_K, nullptr, Kp);
  gemm_f16wmma<false><<<gGemm, 256, 0, stream>>>(vh, Wvh, b_V, nullptr, Vp);
  flash_attn_wmma<<<dim3(S_ / 128, H_, B_), 256, 0, stream>>>(Qp, Kp, Vp, Mg);
  gemm_f16wmma<true><<<gGemm, 256, 0, stream>>>(Mg, Woh, b_O, (float*)d_out, nullptr);
}
